// aggregator_event_DynamicGCN_60988535603562
// MI455X (gfx1250) — matrix operand order repaired and checked by reading, not compiled, not run on hardware
//
#include <hip/hip_runtime.h>
#include <hip/hip_bf16.h>
#include <math.h>

#define VOCAB_N   100000
#define H_DIM     128
#define T_N       10
#define NSUB_N    20000
#define E_N       320000
#define DTEXT_N   768
#define BN_EPS    1e-5f

typedef __bf16 bf16_t;
typedef bf16_t v16bf __attribute__((ext_vector_type(16)));
typedef float  v8f   __attribute__((ext_vector_type(8)));

// ---------------------------------------------------------------------------
// f32 -> bf16 weight conversion (weights are reused by every WMMA GEMM)
// ---------------------------------------------------------------------------
__global__ void k_cvt_f32_bf16(const float* __restrict__ src, bf16_t* __restrict__ dst, int n) {
  int i = blockIdx.x * blockDim.x + threadIdx.x;
  if (i < n) dst[i] = (bf16_t)src[i];
}

__global__ void k_zero_f32(float* __restrict__ p, int n) {
  int i = blockIdx.x * blockDim.x + threadIdx.x;
  if (i < n) p[i] = 0.0f;
}

// ---------------------------------------------------------------------------
// WMMA GEMM:  C[rows x 128] = act( A[rows x K](f32) @ W[K x 128](bf16) + bias )
// optionally accumulating into existing C.  One wave -> 16 rows x 128 cols.
// rows % 16 == 0, K % 32 == 0.  Fragment layouts per CDNA5 ISA 7.12.2.
// ---------------------------------------------------------------------------
__global__ void k_gemm_wmma_bf16_n128(const float* __restrict__ A,
                                      const bf16_t* __restrict__ W,
                                      const float* __restrict__ bias,
                                      float* __restrict__ C,
                                      int rows, int K, int act, int accum) {
  const int N = 128;
  const int wave = threadIdx.x >> 5;
  const int lane = threadIdx.x & 31;
  const int lo = lane & 15;
  const int hi = lane >> 4;
  const int rowBlock = (blockIdx.x * (blockDim.x >> 5) + wave) << 4;
  if (rowBlock >= rows) return;          // wave-uniform: EXEC stays all-ones

  v8f acc[8];
  if (accum) {
#pragma unroll
    for (int nt = 0; nt < 8; ++nt)
#pragma unroll
      for (int v = 0; v < 8; ++v)
        acc[nt][v] = C[(size_t)(rowBlock + v + hi * 8) * N + nt * 16 + lo];
  } else {
#pragma unroll
    for (int nt = 0; nt < 8; ++nt)
#pragma unroll
      for (int v = 0; v < 8; ++v) acc[nt][v] = 0.0f;
  }

  const float* Arow = A + (size_t)(rowBlock + lo) * K;
  for (int kc = 0; kc < K; kc += 32) {
    // A fragment 16x32 bf16: lane lo = row; vgpr v holds K pair
    v16bf a;
#pragma unroll
    for (int v = 0; v < 8; ++v) {
      int kb = (v < 4 ? 2 * v : 16 + 2 * (v - 4)) + hi * 8 + kc;
      a[2 * v]     = (bf16_t)Arow[kb];
      a[2 * v + 1] = (bf16_t)Arow[kb + 1];
    }
    // B fragment 32x16 bf16 per n-tile: lane carries N, vgpr j carries K pair
    const bf16_t* Wrow = W + (size_t)(kc + hi * 8) * N + lo;
#pragma unroll
    for (int nt = 0; nt < 8; ++nt) {
      v16bf b;
#pragma unroll
      for (int j = 0; j < 8; ++j) {
        b[2 * j]     = Wrow[(size_t)(j < 4 ? 2 * j : 16 + 2 * (j - 4)) * N + nt * 16];
        b[2 * j + 1] = Wrow[(size_t)((j < 4 ? 2 * j : 16 + 2 * (j - 4)) + 1) * N + nt * 16];
      }
      acc[nt] = __builtin_amdgcn_wmma_f32_16x16x32_bf16(
          false, a, false, b, (short)0, acc[nt], false, false);
    }
  }

#pragma unroll
  for (int nt = 0; nt < 8; ++nt)
#pragma unroll
    for (int v = 0; v < 8; ++v) {
      int row = rowBlock + v + hi * 8;
      int col = nt * 16 + lo;
      float x = acc[nt][v] + (bias ? bias[col] : 0.0f);
      if (act) x = fmaxf(x, 0.0f);
      C[(size_t)row * N + col] = x;
    }
}

// ---------------------------------------------------------------------------
// Graph kernels (bandwidth-bound; float4 coalesced, HW float atomics)
// ---------------------------------------------------------------------------
__global__ void k_gather_rows(const float* __restrict__ src, const int* __restrict__ nid,
                              float* __restrict__ dst, int nrows) {
  int gid = blockIdx.x * blockDim.x + threadIdx.x;
  int row = gid >> 5;
  int f = (gid & 31) << 2;
  if (row >= nrows) return;
  const float4 v = *(const float4*)(src + (size_t)nid[row] * H_DIM + f);
  *(float4*)(dst + (size_t)row * H_DIM + f) = v;
}

__global__ void k_scatter_rows(const float* __restrict__ src, const int* __restrict__ nid,
                               float* __restrict__ dst, int nrows) {
  int gid = blockIdx.x * blockDim.x + threadIdx.x;
  int row = gid >> 5;
  int f = (gid & 31) << 2;
  if (row >= nrows) return;
  const float4 v = *(const float4*)(src + (size_t)row * H_DIM + f);
  *(float4*)(dst + (size_t)nid[row] * H_DIM + f) = v;
}

__global__ void k_degrees(const int* __restrict__ src, const int* __restrict__ dst,
                          float* __restrict__ deg_out, float* __restrict__ deg_in, int ne) {
  int e = blockIdx.x * blockDim.x + threadIdx.x;
  if (e >= ne) return;
  atomicAdd(deg_out + src[e], 1.0f);
  atomicAdd(deg_in + dst[e], 1.0f);
}

__global__ void k_edge_messages(const float* __restrict__ h,
                                const int* __restrict__ src, const int* __restrict__ dst,
                                const float* __restrict__ deg_out, const float* __restrict__ deg_in,
                                float* __restrict__ agg, int ne) {
  unsigned gid = blockIdx.x * blockDim.x + threadIdx.x;
  int e = gid >> 5;
  int f = (gid & 31) << 2;
  if (e >= ne) return;
  int s = src[e], d = dst[e];
  if (e + 4096 < ne)  // lowers to global_prefetch_b8 on gfx1250
    __builtin_prefetch(h + (size_t)src[e + 4096] * H_DIM + f, 0, 1);
  float norm = rsqrtf(fmaxf(deg_out[s], 1.0f)) * rsqrtf(fmaxf(deg_in[d], 1.0f));
  const float4 v = *(const float4*)(h + (size_t)s * H_DIM + f);
  float* ap = agg + (size_t)d * H_DIM + f;
  atomicAdd(ap + 0, v.x * norm);
  atomicAdd(ap + 1, v.y * norm);
  atomicAdd(ap + 2, v.z * norm);
  atomicAdd(ap + 3, v.w * norm);
}

// ---------------------------------------------------------------------------
// BatchNorm (population stats), GRU combine, last-layer matvec, final reduce
// ---------------------------------------------------------------------------
__global__ void k_bn_stats(const float* __restrict__ X, int rows, int cols,
                           float* __restrict__ mean, float* __restrict__ var) {
  int c = blockIdx.x;
  __shared__ float s1[256], s2[256];
  float s = 0.0f, q = 0.0f;
  for (int r = threadIdx.x; r < rows; r += blockDim.x) {
    float v = X[(size_t)r * cols + c];
    s += v; q += v * v;
  }
  s1[threadIdx.x] = s; s2[threadIdx.x] = q;
  __syncthreads();
  for (int off = blockDim.x >> 1; off > 0; off >>= 1) {
    if (threadIdx.x < off) { s1[threadIdx.x] += s1[threadIdx.x + off]; s2[threadIdx.x] += s2[threadIdx.x + off]; }
    __syncthreads();
  }
  if (threadIdx.x == 0) {
    float m = s1[0] / rows;
    mean[c] = m;
    var[c] = s2[0] / rows - m * m;
  }
}

__global__ void k_bn_apply_relu(float* __restrict__ X, const float* __restrict__ mean,
                                const float* __restrict__ var, const float* __restrict__ g,
                                const float* __restrict__ b, int rows, int cols) {
  int i = blockIdx.x * blockDim.x + threadIdx.x;
  if (i >= rows * cols) return;
  int c = i % cols;
  float x = (X[i] - mean[c]) * rsqrtf(var[c] + BN_EPS) * g[c] + b[c];
  X[i] = fmaxf(x, 0.0f);
}

__global__ void k_temporal_combine(const float* __restrict__ tr, const float* __restrict__ tz,
                                   const float* __restrict__ txn, const float* __restrict__ thn,
                                   const float* __restrict__ h0, float* __restrict__ out, int n) {
  int i = blockIdx.x * blockDim.x + threadIdx.x;
  if (i >= n) return;
  float r = 1.0f / (1.0f + expf(-tr[i]));
  float z = 1.0f / (1.0f + expf(-tz[i]));
  float nn = tanhf(txn[i] + r * thn[i]);
  out[i] = (1.0f - z) * nn + z * h0[i];
}

__global__ void k_matvec_add_relu(const float* __restrict__ A, const float* __restrict__ w,
                                  const float* __restrict__ b, float* __restrict__ out, int rows) {
  int r = blockIdx.x * blockDim.x + threadIdx.x;
  if (r >= rows) return;
  float s = b[0];
  const float* ar = A + (size_t)r * H_DIM;
#pragma unroll 8
  for (int k = 0; k < H_DIM; ++k) s += ar[k] * w[k];
  out[r] = fmaxf(s, 0.0f);
}

__global__ void k_masked_dot(const float* __restrict__ xL, const int* __restrict__ nid,
                             const float* __restrict__ maskW, float* __restrict__ accum, int n) {
  __shared__ float sh[256];
  int tid = threadIdx.x;
  float s = 0.0f;
  for (int i = blockIdx.x * blockDim.x + tid; i < n; i += gridDim.x * blockDim.x)
    s += xL[i] * maskW[nid[i]];
  sh[tid] = s;
  __syncthreads();
  for (int off = blockDim.x >> 1; off > 0; off >>= 1) {
    if (tid < off) sh[tid] += sh[tid + off];
    __syncthreads();
  }
  if (tid == 0) atomicAdd(accum, sh[0]);
}

__global__ void k_finalize_sigmoid(const float* __restrict__ accum, const float* __restrict__ mask_b,
                                   float* __restrict__ out) {
  if (blockIdx.x == 0 && threadIdx.x == 0) {
    float v = accum[0] + mask_b[0];
    out[0] = 1.0f / (1.0f + expf(-v));
  }
}

// ---------------------------------------------------------------------------
// Host orchestration
// ---------------------------------------------------------------------------
extern "C" void kernel_launch(void* const* d_in, const int* in_sizes, int n_in,
                              void* d_out, int out_size, void* d_ws, size_t ws_size,
                              hipStream_t stream) {
  (void)in_sizes; (void)n_in; (void)out_size; (void)ws_size;

  // setup_inputs() flattening order: word_embeds, params (insertion order), node_ids, edges_src, edges_dst
  const float* word_embeds = (const float*)d_in[0];
  const float* W_text = (const float*)d_in[1];
  const float* b_text = (const float*)d_in[2];
  const float* g1_W1 = (const float*)d_in[3];
  const float* g1_b1 = (const float*)d_in[4];
  const float* g1_W2 = (const float*)d_in[5];
  const float* g1_b2 = (const float*)d_in[6];
  const float* bn_g = (const float*)d_in[7];
  const float* bn_b = (const float*)d_in[8];
  const float* Wr = (const float*)d_in[9];
  const float* Ur = (const float*)d_in[10];
  const float* br = (const float*)d_in[11];
  const float* Wz = (const float*)d_in[12];
  const float* Uz = (const float*)d_in[13];
  const float* bz = (const float*)d_in[14];
  const float* Wn = (const float*)d_in[15];
  const float* Un = (const float*)d_in[16];
  const float* bn_t = (const float*)d_in[17];
  const float* gL_W1 = (const float*)d_in[18];
  const float* gL_b1 = (const float*)d_in[19];
  const float* gL_W2 = (const float*)d_in[20];
  const float* gL_b2 = (const float*)d_in[21];
  const float* bnL_g = (const float*)d_in[22];
  const float* bnL_b = (const float*)d_in[23];
  const float* mask_W = (const float*)d_in[24];
  const float* mask_b = (const float*)d_in[25];
  const int* node_ids = (const int*)d_in[26];
  const int* edges_src = (const int*)d_in[27];
  const int* edges_dst = (const int*)d_in[28];

  // workspace carve-up (256B aligned chunks)
  char* ws = (char*)d_ws;
  size_t o = 0;
  auto take = [&](size_t bytes) -> char* {
    char* p = ws + o;
    o += (bytes + 255) & ~(size_t)255;
    return p;
  };
  float* last_x = (float*)take((size_t)VOCAB_N * H_DIM * 4);
  float* B0 = (float*)take((size_t)NSUB_N * H_DIM * 4);  // x0 (prev state gathered)
  float* B1 = (float*)take((size_t)NSUB_N * H_DIM * 4);  // agg scratch / tr
  float* B2 = (float*)take((size_t)NSUB_N * H_DIM * 4);  // x1 / tz
  float* B3 = (float*)take((size_t)NSUB_N * H_DIM * 4);  // x2 / gru out
  float* B4 = (float*)take((size_t)NSUB_N * H_DIM * 4);  // txn
  float* B5 = (float*)take((size_t)NSUB_N * H_DIM * 4);  // thn
  bf16_t* Wb_text = (bf16_t*)take((size_t)DTEXT_N * H_DIM * 2);
  bf16_t* Wb_g1W1 = (bf16_t*)take((size_t)H_DIM * H_DIM * 2);
  bf16_t* Wb_g1W2 = (bf16_t*)take((size_t)H_DIM * H_DIM * 2);
  bf16_t* Wb_Wr = (bf16_t*)take((size_t)H_DIM * H_DIM * 2);
  bf16_t* Wb_Ur = (bf16_t*)take((size_t)H_DIM * H_DIM * 2);
  bf16_t* Wb_Wz = (bf16_t*)take((size_t)H_DIM * H_DIM * 2);
  bf16_t* Wb_Uz = (bf16_t*)take((size_t)H_DIM * H_DIM * 2);
  bf16_t* Wb_Wn = (bf16_t*)take((size_t)H_DIM * H_DIM * 2);
  bf16_t* Wb_Un = (bf16_t*)take((size_t)H_DIM * H_DIM * 2);
  bf16_t* Wb_gLW1 = (bf16_t*)take((size_t)H_DIM * H_DIM * 2);
  float* deg_out = (float*)take((size_t)NSUB_N * 4);
  float* deg_in = (float*)take((size_t)NSUB_N * 4);
  float* bn_mean = (float*)take(H_DIM * 4);
  float* bn_var = (float*)take(H_DIM * 4);
  float* xL = (float*)take((size_t)NSUB_N * 4);
  float* red = (float*)take(256);

  auto cvt = [&](const float* s, bf16_t* d, int n) {
    k_cvt_f32_bf16<<<(n + 255) / 256, 256, 0, stream>>>(s, d, n);
  };
  auto zero = [&](float* p, int n) {
    k_zero_f32<<<(n + 255) / 256, 256, 0, stream>>>(p, n);
  };
  auto gemm = [&](const float* A, const bf16_t* W, const float* bias, float* C,
                  int rows, int K, int act, int accum) {
    int rowBlocks = rows / 16;          // rows always multiple of 16 here
    const int wavesPerBlock = 4;
    dim3 grid((rowBlocks + wavesPerBlock - 1) / wavesPerBlock);
    k_gemm_wmma_bf16_n128<<<grid, 32 * wavesPerBlock, 0, stream>>>(A, W, bias, C, rows, K, act, accum);
  };

  // ---- weight conversions (every launch; deterministic) ----
  cvt(W_text, Wb_text, DTEXT_N * H_DIM);
  cvt(g1_W1, Wb_g1W1, H_DIM * H_DIM);
  cvt(g1_W2, Wb_g1W2, H_DIM * H_DIM);
  cvt(Wr, Wb_Wr, H_DIM * H_DIM);
  cvt(Ur, Wb_Ur, H_DIM * H_DIM);
  cvt(Wz, Wb_Wz, H_DIM * H_DIM);
  cvt(Uz, Wb_Uz, H_DIM * H_DIM);
  cvt(Wn, Wb_Wn, H_DIM * H_DIM);
  cvt(Un, Wb_Un, H_DIM * H_DIM);
  cvt(gL_W1, Wb_gLW1, H_DIM * H_DIM);

  // ---- text projection: last_x = word_embeds @ W_text + b_text ----
  gemm(word_embeds, Wb_text, b_text, last_x, VOCAB_N, DTEXT_N, /*act=*/0, /*accum=*/0);

  const int gatherThreads = NSUB_N * 32;
  const int edgeThreads = E_N * 32;
  const int nh = NSUB_N * H_DIM;

  for (int t = 0; t < T_N; ++t) {
    const int* nid = node_ids + (size_t)t * NSUB_N;
    const int* src = edges_src + (size_t)t * E_N;
    const int* dst = edges_dst + (size_t)t * E_N;

    k_gather_rows<<<(gatherThreads + 255) / 256, 256, 0, stream>>>(last_x, nid, B0, NSUB_N);

    zero(deg_out, NSUB_N);
    zero(deg_in, NSUB_N);
    k_degrees<<<(E_N + 255) / 256, 256, 0, stream>>>(src, dst, deg_out, deg_in, E_N);

    if (t < T_N - 1) {
      // GCN layer 1: agg(x0) -> relu(@W1 + b1)
      zero(B1, nh);
      k_edge_messages<<<(edgeThreads + 255) / 256, 256, 0, stream>>>(B0, src, dst, deg_out, deg_in, B1, E_N);
      gemm(B1, Wb_g1W1, g1_b1, B2, NSUB_N, H_DIM, 1, 0);
      // GCN layer 2
      zero(B1, nh);
      k_edge_messages<<<(edgeThreads + 255) / 256, 256, 0, stream>>>(B2, src, dst, deg_out, deg_in, B1, E_N);
      gemm(B1, Wb_g1W2, g1_b2, B3, NSUB_N, H_DIM, 1, 0);
      // BN + relu
      k_bn_stats<<<H_DIM, 256, 0, stream>>>(B3, NSUB_N, H_DIM, bn_mean, bn_var);
      k_bn_apply_relu<<<(nh + 255) / 256, 256, 0, stream>>>(B3, bn_mean, bn_var, bn_g, bn_b, NSUB_N, H_DIM);
      // temporal GRU: x = B3, h = B0
      gemm(B3, Wb_Wr, br, B1, NSUB_N, H_DIM, 0, 0);
      gemm(B0, Wb_Ur, nullptr, B1, NSUB_N, H_DIM, 0, 1);   // tr = xWr+br+hUr
      gemm(B3, Wb_Wz, bz, B2, NSUB_N, H_DIM, 0, 0);
      gemm(B0, Wb_Uz, nullptr, B2, NSUB_N, H_DIM, 0, 1);   // tz
      gemm(B3, Wb_Wn, bn_t, B4, NSUB_N, H_DIM, 0, 0);      // txn
      gemm(B0, Wb_Un, nullptr, B5, NSUB_N, H_DIM, 0, 0);   // thn
      k_temporal_combine<<<(nh + 255) / 256, 256, 0, stream>>>(B1, B2, B4, B5, B0, B3, nh);
      k_scatter_rows<<<(gatherThreads + 255) / 256, 256, 0, stream>>>(B3, nid, last_x, NSUB_N);
    } else {
      // last-layer GCN: h -> h -> OUT(=1)
      zero(B1, nh);
      k_edge_messages<<<(edgeThreads + 255) / 256, 256, 0, stream>>>(B0, src, dst, deg_out, deg_in, B1, E_N);
      gemm(B1, Wb_gLW1, gL_b1, B2, NSUB_N, H_DIM, 1, 0);
      zero(B1, nh);
      k_edge_messages<<<(edgeThreads + 255) / 256, 256, 0, stream>>>(B2, src, dst, deg_out, deg_in, B1, E_N);
      k_matvec_add_relu<<<(NSUB_N + 255) / 256, 256, 0, stream>>>(B1, gL_W2, gL_b2, xL, NSUB_N);
      // BN over single output column, then relu
      k_bn_stats<<<1, 256, 0, stream>>>(xL, NSUB_N, 1, bn_mean, bn_var);
      k_bn_apply_relu<<<(NSUB_N + 255) / 256, 256, 0, stream>>>(xL, bn_mean, bn_var, bnL_g, bnL_b, NSUB_N, 1);
      // sigmoid(full @ mask_W + mask_b): full is zero off-subgraph, so reduce over nid
      zero(red, 1);
      k_masked_dot<<<256, 256, 0, stream>>>(xL, nid, mask_W, red, NSUB_N);
      k_finalize_sigmoid<<<1, 1, 0, stream>>>(red, mask_b, (float*)d_out);
    }
  }
}
